// _AttentionLayer_22419729285704
// MI455X (gfx1250) — matrix operand order repaired and checked by reading, not compiled, not run on hardware
//
#include <hip/hip_runtime.h>

#define S_LEN    2048
#define DM       1024
#define N_HEADS  16
#define HEAD_DIM 64
#define QKV_COLS 3072

typedef __attribute__((ext_vector_type(16))) _Float16 v16h;
typedef __attribute__((ext_vector_type(8)))  _Float16 v8h;
typedef __attribute__((ext_vector_type(8)))  float    v8f;

static __device__ __forceinline__ v16h cat8(v8h lo, v8h hi) {
  return __builtin_shufflevector(lo, hi, 0,1,2,3,4,5,6,7,8,9,10,11,12,13,14,15);
}

static __device__ __forceinline__ v8f wmma_f16(v16h a, v16h b, v8f c) {
  // v_wmma_f32_16x16x32_f16: (neg_a, A, neg_b, B, c_mod, C, reuse_a, reuse_b)
  return __builtin_amdgcn_wmma_f32_16x16x32_f16(false, a, false, b, (short)0, c, false, false);
}

// A fragment (16x32 f16, M x K): lane holds row, K chunk kb..kb+15,
// kb = 16*(lane>>4). Row-major source, leading dim ldk.
static __device__ __forceinline__ v16h load_a_frag(const _Float16* __restrict__ base,
                                                   int row, int ldk, int k0, int lane) {
  const int kb = (lane >> 4) * 16;
  const _Float16* p = base + (size_t)row * ldk + k0 + kb;
  v8h lo = *(const v8h*)p;
  v8h hi = *(const v8h*)(p + 8);
  return cat8(lo, hi);
}

// B fragment (32x16 f16, K x N) where memory holds B^T row-major (weight row n is
// K-contiguous): lane needs B[kb16+i][n] = W[n][k0+kb16+i], 16 contiguous halfs (32B).
static __device__ __forceinline__ v16h load_b_frag_kcontig(const _Float16* __restrict__ base,
                                                           int nrow, int ldk, int k0, int lane) {
  const int kb16 = (lane >> 4) * 16;
  return *(const v16h*)(base + (size_t)nrow * ldk + k0 + kb16);
}

// ---------------------------------------------------------------- casts
__global__ __launch_bounds__(256) void cast_f32_f16(const float* __restrict__ src,
                                                    _Float16* __restrict__ dst, int n) {
  int i = blockIdx.x * blockDim.x + threadIdx.x;
  if (i < n) dst[i] = (_Float16)src[i];
}

// ---------------------------------------------------------------- QKV projection
// C[S,3072] = h16 @ Wqkv^T + b_qkv. One 16x64 strip per wave (4 accumulators, 4x A reuse).
// Q,K sections stored row-major into qkv16[S][3072]; V section stored TRANSPOSED into
// vt16[1024][S] so attention's V B-fragments become contiguous 32B loads.
__global__ __launch_bounds__(128) void qkv_gemm_kernel(const _Float16* __restrict__ h16,
                                                       const _Float16* __restrict__ w16,
                                                       const float* __restrict__ bias,
                                                       _Float16* __restrict__ qkv16,
                                                       _Float16* __restrict__ vt16) {
  const int lane = threadIdx.x & 31;
  const int wave = threadIdx.x >> 5;
  const int ngroups = QKV_COLS / 64;              // 48 strips of 4 N-tiles
  const int task = blockIdx.x * 4 + wave;         // 128*48 = 6144 tasks
  const int tm = task / ngroups;
  const int tg = task % ngroups;
  const int l15 = lane & 15;
  const int rowA = tm * 16 + l15;

  v8f acc[4] = {};
  for (int k0 = 0; k0 < DM; k0 += 32) {
    v16h a = load_a_frag(h16, rowA, DM, k0, lane);
#pragma unroll
    for (int t = 0; t < 4; ++t) {
      const int rowB = (tg * 4 + t) * 16 + l15;
      __builtin_prefetch(w16 + (size_t)rowB * DM + k0 + 256, 0, 1);
      v16h b = load_b_frag_kcontig(w16, rowB, DM, k0, lane);
      acc[t] = wmma_f16(a, b, acc[t]);
    }
  }
  const int rbase = tm * 16 + (lane >> 4) * 8;
#pragma unroll
  for (int t = 0; t < 4; ++t) {
    const int gcol = (tg * 4 + t) * 16 + l15;     // C-layout: col fixed per lane
    const float bs = bias[gcol];
    if (gcol < 2 * DM) {                          // Q and K sections: row-major
#pragma unroll
      for (int i = 0; i < 8; ++i)
        qkv16[(size_t)(rbase + i) * QKV_COLS + gcol] = (_Float16)(acc[t][i] + bs);
    } else {                                      // V section: transposed [d][key]
      _Float16* vrow = vt16 + (size_t)(gcol - 2 * DM) * S_LEN + rbase;
#pragma unroll
      for (int i = 0; i < 8; ++i)                 // consecutive keys -> coalescable
        vrow[i] = (_Float16)(acc[t][i] + bs);
    }
  }
}

// ---------------------------------------------------------------- flash attention
// 4 waves/block, each wave owns one (query 16-tile, head). Online softmax, 32 keys/step.
__global__ __launch_bounds__(128) void attn_kernel(const _Float16* __restrict__ qkv16,
                                                   const _Float16* __restrict__ vt16,
                                                   _Float16* __restrict__ attn16) {
  __shared__ _Float16 Plds[4 * 16 * 32];          // 1KB per wave
  const int lane = threadIdx.x & 31;
  const int wave = threadIdx.x >> 5;
  const int qt   = blockIdx.x * 4 + wave;         // 0..127
  const int head = blockIdx.y;                    // 0..15
  const int l15  = lane & 15;
  const int hih  = lane >> 4;
  _Float16* P = Plds + wave * 512;

  // Q fragments (16x64 split into two K=32 steps), reused across all key blocks
  const _Float16* qbase = qkv16 + (size_t)(qt * 16 + l15) * QKV_COLS + head * HEAD_DIM;
  v16h aq0, aq1;
  {
    const int kb = hih * 16;
    aq0 = cat8(*(const v8h*)(qbase + kb),      *(const v8h*)(qbase + kb + 8));
    aq1 = cat8(*(const v8h*)(qbase + 32 + kb), *(const v8h*)(qbase + 32 + kb + 8));
  }

  float m[8], l[8];
#pragma unroll
  for (int i = 0; i < 8; ++i) { m[i] = -1e30f; l[i] = 0.0f; }
  v8f o[4] = {};

  const float scale = 0.125f;                     // 1/sqrt(64)
  const int kb16 = hih * 16;

  for (int jb = 0; jb < S_LEN; jb += 32) {
    // scores: two 16x16 tiles (keys jb..+15, jb+16..+31)
    v8f s0 = {}, s1 = {};
#pragma unroll
    for (int kk = 0; kk < 2; ++kk) {
      const int k0 = kk * 32;
      const _Float16* kcol = qkv16 + DM + head * HEAD_DIM + k0 + kb16;
      v16h b0 = *(const v16h*)(kcol + (size_t)(jb + l15) * QKV_COLS);
      v16h b1 = *(const v16h*)(kcol + (size_t)(jb + 16 + l15) * QKV_COLS);
      v16h aq = kk ? aq1 : aq0;
      s0 = wmma_f16(aq, b0, s0);
      s1 = wmma_f16(aq, b1, s1);
    }
    // online softmax over 32 columns; row stats replicated across each 16-lane group
#pragma unroll
    for (int i = 0; i < 8; ++i) {
      float v0 = s0[i] * scale;
      float v1 = s1[i] * scale;
      float mx = fmaxf(v0, v1);
      mx = fmaxf(mx, __shfl_xor(mx, 1));
      mx = fmaxf(mx, __shfl_xor(mx, 2));
      mx = fmaxf(mx, __shfl_xor(mx, 4));
      mx = fmaxf(mx, __shfl_xor(mx, 8));
      float newm = fmaxf(m[i], mx);
      float p0 = __expf(v0 - newm);
      float p1 = __expf(v1 - newm);
      float corr = __expf(m[i] - newm);
      float ps = p0 + p1;
      ps += __shfl_xor(ps, 1);
      ps += __shfl_xor(ps, 2);
      ps += __shfl_xor(ps, 4);
      ps += __shfl_xor(ps, 8);
      l[i] = l[i] * corr + ps;
      m[i] = newm;
#pragma unroll
      for (int t = 0; t < 4; ++t) o[t][i] *= corr;
      const int r = hih * 8 + i;                  // C-layout row
      P[r * 32 + l15]      = (_Float16)p0;
      P[r * 32 + 16 + l15] = (_Float16)p1;
    }
    __syncthreads();
    // reload P as an A-layout 16x32 f16 fragment
    v16h ap;
    {
      const int kb = hih * 16;
      const _Float16* pp = P + l15 * 32 + kb;
      ap = cat8(*(const v8h*)pp, *(const v8h*)(pp + 8));
    }
    // O(16x64) += P(16x32) @ V_block(32x64); V^T layout -> contiguous B fragments
#pragma unroll
    for (int t = 0; t < 4; ++t) {
      const _Float16* vrow = vt16 + (size_t)(head * HEAD_DIM + t * 16 + l15) * S_LEN;
      v16h bv = *(const v16h*)(vrow + jb + kb16);
      o[t] = wmma_f16(ap, bv, o[t]);
    }
    __syncthreads();
  }

  const int rbase = qt * 16 + hih * 8;
#pragma unroll
  for (int i = 0; i < 8; ++i) {
    const float inv = 1.0f / l[i];
    const size_t row = rbase + i;
#pragma unroll
    for (int t = 0; t < 4; ++t)
      attn16[row * DM + head * HEAD_DIM + t * 16 + l15] = (_Float16)(o[t][i] * inv);
  }
}

// ---------------------------------------------------------------- out-proj + residual + LN
// One block per 16-row stripe; 8 waves, each computing 16x64 strips (4x A reuse),
// rows staged in dynamic LDS (fp32), then fused LayerNorm.
__global__ __launch_bounds__(256) void outproj_ln_kernel(const _Float16* __restrict__ attn16,
                                                         const _Float16* __restrict__ wout16,
                                                         const float* __restrict__ b_out,
                                                         const float* __restrict__ h,
                                                         const float* __restrict__ gamma,
                                                         const float* __restrict__ beta,
                                                         float* __restrict__ out) {
  extern __shared__ float smem[];
  float* xbuf = smem;                    // 16 x 1024
  float* rsum = smem + 16 * DM;          // 16
  float* rsq  = rsum + 16;               // 16
  float* mu   = rsq + 16;                // 16
  float* rstd = mu + 16;                 // 16

  const int lane = threadIdx.x & 31;
  const int wave = threadIdx.x >> 5;
  const int stripe = blockIdx.x;
  const int l15 = lane & 15;
  const int hih = lane >> 4;
  const int rowA = stripe * 16 + l15;

  if (threadIdx.x < 16) { rsum[threadIdx.x] = 0.0f; rsq[threadIdx.x] = 0.0f; }

  for (int ng = wave; ng < DM / 64; ng += 8) {    // 16 strip-groups, 2 per wave
    v8f acc[4] = {};
    for (int k0 = 0; k0 < DM; k0 += 32) {
      v16h a = load_a_frag(attn16, rowA, DM, k0, lane);
#pragma unroll
      for (int t = 0; t < 4; ++t) {
        const int rowB = (ng * 4 + t) * 16 + l15;
        __builtin_prefetch(wout16 + (size_t)rowB * DM + k0 + 256, 0, 1);
        v16h b = load_b_frag_kcontig(wout16, rowB, DM, k0, lane);
        acc[t] = wmma_f16(a, b, acc[t]);
      }
    }
#pragma unroll
    for (int t = 0; t < 4; ++t) {
      const int gcol = (ng * 4 + t) * 16 + l15;
      const float bs = b_out[gcol];
#pragma unroll
      for (int i = 0; i < 8; ++i) {
        const int r = hih * 8 + i;
        const int grow = stripe * 16 + r;
        xbuf[r * DM + gcol] = acc[t][i] + bs + h[(size_t)grow * DM + gcol];
      }
    }
  }
  __syncthreads();

  // LayerNorm: 16 threads per row
  const int row = threadIdx.x >> 4;
  const int t16 = threadIdx.x & 15;
  float s = 0.0f, sq = 0.0f;
  for (int c = t16; c < DM; c += 16) {
    const float x = xbuf[row * DM + c];
    s += x; sq += x * x;
  }
  atomicAdd(&rsum[row], s);
  atomicAdd(&rsq[row], sq);
  __syncthreads();
  if (threadIdx.x < 16) {
    const float mean = rsum[threadIdx.x] * (1.0f / DM);
    const float var  = rsq[threadIdx.x] * (1.0f / DM) - mean * mean;
    mu[threadIdx.x]   = mean;
    rstd[threadIdx.x] = rsqrtf(var + 1e-5f);
  }
  __syncthreads();
  const int grow = stripe * 16 + row;
  for (int c = t16; c < DM; c += 16) {
    const float x = xbuf[row * DM + c];
    out[(size_t)grow * DM + c] = (x - mu[row]) * rstd[row] * gamma[c] + beta[c];
  }
}

// ---------------------------------------------------------------- launch
extern "C" void kernel_launch(void* const* d_in, const int* in_sizes, int n_in,
                              void* d_out, int out_size, void* d_ws, size_t ws_size,
                              hipStream_t stream) {
  const float* h     = (const float*)d_in[0];
  const float* W_qkv = (const float*)d_in[1];
  const float* b_qkv = (const float*)d_in[2];
  const float* W_out = (const float*)d_in[3];
  const float* b_out = (const float*)d_in[4];
  const float* gamma = (const float*)d_in[5];
  const float* beta  = (const float*)d_in[6];
  float* out = (float*)d_out;

  char* ws = (char*)d_ws;
  _Float16* h16    = (_Float16*)ws; ws += (size_t)S_LEN * DM * 2;        // 4 MB
  _Float16* wqkv16 = (_Float16*)ws; ws += (size_t)3 * DM * DM * 2;       // 6 MB
  _Float16* wout16 = (_Float16*)ws; ws += (size_t)DM * DM * 2;           // 2 MB
  _Float16* qkv16  = (_Float16*)ws; ws += (size_t)S_LEN * QKV_COLS * 2;  // 12 MB (Q,K)
  _Float16* vt16   = (_Float16*)ws; ws += (size_t)DM * S_LEN * 2;        // 4 MB (V^T)
  _Float16* attn16 = (_Float16*)ws; ws += (size_t)S_LEN * DM * 2;        // 4 MB

  const int nh  = S_LEN * DM;
  const int nwq = 3 * DM * DM;
  const int nwo = DM * DM;
  cast_f32_f16<<<(nh  + 255) / 256, 256, 0, stream>>>(h,     h16,    nh);
  cast_f32_f16<<<(nwq + 255) / 256, 256, 0, stream>>>(W_qkv, wqkv16, nwq);
  cast_f32_f16<<<(nwo + 255) / 256, 256, 0, stream>>>(W_out, wout16, nwo);

  const int qkv_tasks = (S_LEN / 16) * (QKV_COLS / 64);   // 6144 strips
  qkv_gemm_kernel<<<qkv_tasks / 4, 128, 0, stream>>>(h16, wqkv16, b_qkv, qkv16, vt16);

  attn_kernel<<<dim3(S_LEN / 16 / 4, N_HEADS), 128, 0, stream>>>(qkv16, vt16, attn16);

  const size_t lnshm = (size_t)(16 * DM + 64) * sizeof(float);
  outproj_ln_kernel<<<S_LEN / 16, 256, lnshm, stream>>>(attn16, wout16, b_out, h,
                                                        gamma, beta, out);
}
